// SrcGNN_58712202936407
// MI455X (gfx1250) — matrix operand order repaired and checked by reading, not compiled, not run on hardware
//
#include <hip/hip_runtime.h>
#include <stdint.h>

#define D 512
#define FA_EPS 0.1f
#define BM 128
#define BN 64
#define BK 64
#define LDB 72   // padded LDS stride (ushorts): 144B = 36 dwords -> conflict-free b128 frag reads
#define KTILES (D / BK)

typedef __attribute__((ext_vector_type(16))) __bf16 v16bf;
typedef __attribute__((ext_vector_type(8)))  float  v8f;

// round-to-nearest-even f32 -> bf16
static __device__ __forceinline__ unsigned short f2bf(float f) {
    unsigned int u = __float_as_uint(f);
    u += 0x7FFFu + ((u >> 16) & 1u);
    return (unsigned short)(u >> 16);
}
static __device__ __forceinline__ unsigned pk2bf(float lo, float hi) {
    return (unsigned)f2bf(lo) | ((unsigned)f2bf(hi) << 16);
}

// non-returning native f32 atomic add (L2 atomic unit); avoids CAS-loop lowering
static __device__ __forceinline__ void atomic_add_f32(float* p, float v) {
    asm volatile("global_atomic_add_f32 %0, %1, off"
                 :: "v"((unsigned long long)(uintptr_t)p), "v"(v)
                 : "memory");
}

// async global->LDS 16B copy, tracked by ASYNCcnt (no VGPR data round-trip)
static __device__ __forceinline__ void async_b128(unsigned lds_byte_addr,
                                                  const void* gaddr) {
    asm volatile("global_load_async_to_lds_b128 %0, %1, off"
                 :: "v"(lds_byte_addr), "v"((unsigned long long)(uintptr_t)gaddr)
                 : "memory");
}

__global__ void k_fill(float* __restrict__ p, float val, int n) {
    int i = blockIdx.x * blockDim.x + threadIdx.x;
    if (i < n) p[i] = val;
}

__global__ void k_deg(const long long* __restrict__ dst, float* __restrict__ deg, int E) {
    int e = blockIdx.x * blockDim.x + threadIdx.x;
    if (e < E) atomic_add_f32(&deg[dst[e]], 1.0f);
}

__global__ void k_rsqrt_inplace(float* __restrict__ p, int n) {
    int i = blockIdx.x * blockDim.x + threadIdx.x;
    if (i < n) p[i] = rsqrtf(p[i]);   // deg >= 1 always (self loop)
}

// W[k][n] (f32) -> Wt[n][k] (bf16), so the WMMA B tile is a row-major read
__global__ void k_prep_w(const float* __restrict__ W, unsigned short* __restrict__ Wt) {
    int idx = blockIdx.x * blockDim.x + threadIdx.x;   // over D*D
    int n = idx >> 9, k = idx & (D - 1);
    Wt[idx] = f2bf(W[k * D + n]);
}

// f32 -> bf16 bulk convert (8 elements / thread), keeps conversion out of the GEMM
__global__ void k_cvt_bf16(const float* __restrict__ src, unsigned short* __restrict__ dst,
                           int nchunks) {
    int i = blockIdx.x * blockDim.x + threadIdx.x;
    if (i >= nchunks) return;
    const float4* s = (const float4*)src + (size_t)i * 2;
    float4 a = s[0], b = s[1];
    uint4 o;
    o.x = pk2bf(a.x, a.y);
    o.y = pk2bf(a.z, a.w);
    o.z = pk2bf(b.x, b.y);
    o.w = pk2bf(b.z, b.w);
    ((uint4*)dst)[i] = o;
}

// One wave per node: alpha_l = al.h, alpha_r = ar.h (shuffle reduce), then
// agg[i,:] = h[i,:] * (tanh(al+ar)*dinv^2 + EPS)   (self-loop msg + EPS*x0 fused)
__global__ __launch_bounds__(256) void k_alpha_init(
        const float* __restrict__ h, const float* __restrict__ dinv,
        const float* __restrict__ al, const float* __restrict__ ar,
        float* __restrict__ alpha_l, float* __restrict__ alpha_r,
        float* __restrict__ agg, int n) {
    int wave = (blockIdx.x * blockDim.x + threadIdx.x) >> 5;
    int lane = threadIdx.x & 31;
    if (wave >= n) return;
    const float4* hrow = (const float4*)(h + (size_t)wave * D);
    const float4* a4 = (const float4*)al;
    const float4* b4 = (const float4*)ar;
    float4 hv[4];
    float s = 0.f, t = 0.f;
#pragma unroll
    for (int j = 0; j < 4; ++j) {
        int idx = j * 32 + lane;              // 128 float4 = 512 floats, coalesced
        hv[j] = hrow[idx];
        float4 a = a4[idx], b = b4[idx];
        s += hv[j].x * a.x + hv[j].y * a.y + hv[j].z * a.z + hv[j].w * a.w;
        t += hv[j].x * b.x + hv[j].y * b.y + hv[j].z * b.z + hv[j].w * b.w;
    }
#pragma unroll
    for (int off = 16; off > 0; off >>= 1) {
        s += __shfl_xor(s, off, 32);
        t += __shfl_xor(t, off, 32);
    }
    if (lane == 0) { alpha_l[wave] = s; alpha_r[wave] = t; }
    float di = dinv[wave];
    float coef = tanhf(s + t) * di * di + FA_EPS;
    float4* arow = (float4*)(agg + (size_t)wave * D);
#pragma unroll
    for (int j = 0; j < 4; ++j) {
        int idx = j * 32 + lane;
        float4 v = hv[j];
        v.x *= coef; v.y *= coef; v.z *= coef; v.w *= coef;
        arow[idx] = v;
    }
}

// One wave per edge: agg[dst,:] += coef * h[src,:] via native f32 atomics
__global__ __launch_bounds__(256) void k_edge(
        const float* __restrict__ h,
        const long long* __restrict__ src, const long long* __restrict__ dst,
        const float* __restrict__ alpha_l, const float* __restrict__ alpha_r,
        const float* __restrict__ dinv, float* __restrict__ agg, int E) {
    int e = (blockIdx.x * blockDim.x + threadIdx.x) >> 5;
    int lane = threadIdx.x & 31;
    if (e >= E) return;
    long long s = src[e], d = dst[e];
    float coef = tanhf(alpha_l[s] + alpha_r[d]) * dinv[s] * dinv[d];
    const float4* hrow = (const float4*)(h + (size_t)s * D);
    float* arow = agg + (size_t)d * D;
#pragma unroll
    for (int j = 0; j < 4; ++j) {
        int idx = j * 32 + lane;
        float4 v = hrow[idx];
        int c = idx * 4;
        atomic_add_f32(arow + c + 0, v.x * coef);
        atomic_add_f32(arow + c + 1, v.y * coef);
        atomic_add_f32(arow + c + 2, v.z * coef);
        atomic_add_f32(arow + c + 3, v.w * coef);
    }
}

// C[nrows x 512] = A(bf16) @ W + bias via v_wmma_f32_16x16x32_bf16.
// Double-buffered LDS tiles filled with global_load_async_to_lds_b128 (ASYNCcnt):
// per wave exactly 6 async issues per tile (rows clamped, never predicated), so
// s_wait_asynccnt 6 == "previous tile landed" while the next tile is in flight.
__global__ __launch_bounds__(256) void k_gemm(
        const unsigned short* __restrict__ A, const unsigned short* __restrict__ Wt,
        const float* __restrict__ bias, float* __restrict__ C, int nrows) {
    __shared__ alignas(16) unsigned short lA[2][BM * LDB];   // 2 x 18 KB
    __shared__ alignas(16) unsigned short lB[2][BN * LDB];   // 2 x  9 KB
    int tid = threadIdx.x;
    int wid = tid >> 5, lane = tid & 31;
    int half = lane >> 4, lr = lane & 15;
    int wm = (wid & 3) * 32;
    int wn = (wid >> 2) * 32;
    int blockM = blockIdx.y * BM;
    int blockN = blockIdx.x * BN;

    auto issue_tile = [&](int k0, int buf) {
        // A tile: 128 rows x 64 bf16 = 1024 16B chunks; 4 per thread
#pragma unroll
        for (int it = 0; it < 4; ++it) {
            int c = tid + it * 256;
            int row = c >> 3, q = c & 7;
            int grow = blockM + row;
            if (grow >= nrows) grow = nrows - 1;   // clamp: keeps issue count exact
            unsigned lds = (unsigned)(size_t)(void*)&lA[buf][row * LDB + q * 8];
            async_b128(lds, A + (size_t)grow * D + k0 + q * 8);
        }
        // B tile: 64 rows x 64 bf16 = 512 16B chunks; 2 per thread
#pragma unroll
        for (int it = 0; it < 2; ++it) {
            int c = tid + it * 256;
            int row = c >> 3, q = c & 7;
            unsigned lds = (unsigned)(size_t)(void*)&lB[buf][row * LDB + q * 8];
            async_b128(lds, Wt + (size_t)(blockN + row) * D + k0 + q * 8);
        }
    };

    v8f zero = {0.f, 0.f, 0.f, 0.f, 0.f, 0.f, 0.f, 0.f};
    v8f acc[2][2];
    acc[0][0] = zero; acc[0][1] = zero; acc[1][0] = zero; acc[1][1] = zero;

    issue_tile(0, 0);
    for (int t = 0; t < KTILES; ++t) {
        int buf = t & 1;
        if (t + 1 < KTILES) {
            issue_tile((t + 1) * BK, buf ^ 1);              // prefetch next tile
            asm volatile("s_wait_asynccnt 0x6" ::: "memory"); // tile t landed (in-order)
        } else {
            asm volatile("s_wait_asynccnt 0x0" ::: "memory");
        }
        __syncthreads();   // all waves have tile t in LDS
#pragma unroll
        for (int ks = 0; ks < BK; ks += 32) {
            v16bf afrag[2], bfrag[2];
#pragma unroll
            for (int i = 0; i < 2; ++i) {
                // A 16x32: elems[0..7]=K(half*8+0..7), elems[8..15]=K(16+half*8+0..7)
                const unsigned short* base =
                    &lA[buf][(wm + i * 16 + lr) * LDB + ks + half * 8];
                union { uint4 u[2]; v16bf v; } tmp;
                tmp.u[0] = *(const uint4*)(base);
                tmp.u[1] = *(const uint4*)(base + 16);
                afrag[i] = tmp.v;
            }
#pragma unroll
            for (int j = 0; j < 2; ++j) {
                // B 32x16: N=lr, elems[0..7]=K(half*8+0..7), elems[8..15]=K(16+half*8+0..7)
                const unsigned short* base =
                    &lB[buf][(wn + j * 16 + lr) * LDB + ks + half * 8];
                union { uint4 u[2]; v16bf v; } tmp;
                tmp.u[0] = *(const uint4*)(base);
                tmp.u[1] = *(const uint4*)(base + 16);
                bfrag[j] = tmp.v;
            }
#pragma unroll
            for (int i = 0; i < 2; ++i)
#pragma unroll
                for (int j = 0; j < 2; ++j)
                    acc[i][j] = __builtin_amdgcn_wmma_f32_16x16x32_bf16(
                        false, afrag[i], false, bfrag[j], (short)0, acc[i][j],
                        false, false);
        }
        __syncthreads();   // compute done before next iter overwrites buf
    }
    // C/D layout: lanes 0-15 N=lane, VGPR v -> M=v; lanes 16-31 -> M=8+v
#pragma unroll
    for (int i = 0; i < 2; ++i) {
#pragma unroll
        for (int j = 0; j < 2; ++j) {
            int col = blockN + wn + j * 16 + lr;
            float bv = bias[col];
#pragma unroll
            for (int v = 0; v < 8; ++v) {
                int row = blockM + wm + i * 16 + half * 8 + v;
                if (row < nrows) C[(size_t)row * D + col] = acc[i][j][v] + bv;
            }
        }
    }
}

extern "C" void kernel_launch(void* const* d_in, const int* in_sizes, int n_in,
                              void* d_out, int out_size, void* d_ws, size_t ws_size,
                              hipStream_t stream) {
    const float*     x     = (const float*)d_in[0];
    const long long* edge  = (const long long*)d_in[1];   // int64 [2, E]
    const float*     att_l = (const float*)d_in[2];
    const float*     att_r = (const float*)d_in[3];
    const float*     Ws    = (const float*)d_in[4];
    const float*     bs    = (const float*)d_in[5];
    int n      = in_sizes[0] / D;          // 50000
    int E      = in_sizes[1] / 2;          // 400000
    int layers = in_sizes[4] / (D * D);    // 3

    const long long* srcIdx = edge;
    const long long* dstIdx = edge + E;

    auto align256 = [](size_t v) { return (v + 255) & ~(size_t)255; };
    char* ws = (char*)d_ws;
    float* agg  = (float*)ws;           ws += align256((size_t)n * D * sizeof(float));
    unsigned short* aggbf = (unsigned short*)ws;
                                        ws += align256((size_t)n * D * sizeof(unsigned short));
    float* dinv = (float*)ws;           ws += align256((size_t)n * sizeof(float));
    float* albf = (float*)ws;           ws += align256((size_t)n * sizeof(float));
    float* arbf = (float*)ws;           ws += align256((size_t)n * sizeof(float));
    unsigned short* wt = (unsigned short*)ws;   // D*D bf16

    // degree -> dinv, computed once (graph is static across layers)
    k_fill<<<(n + 255) / 256, 256, 0, stream>>>(dinv, 1.0f, n);               // self loop
    k_deg<<<(E + 255) / 256, 256, 0, stream>>>(dstIdx, dinv, E);
    k_rsqrt_inplace<<<(n + 255) / 256, 256, 0, stream>>>(dinv, n);

    int nchunks = n * D / 8;
    const float* h = x;
    for (int l = 0; l < layers; ++l) {
        k_prep_w<<<(D * D) / 256, 256, 0, stream>>>(Ws + (size_t)l * D * D, wt);
        k_alpha_init<<<(n * 32 + 255) / 256, 256, 0, stream>>>(
            h, dinv, att_l + l * D, att_r + l * D, albf, arbf, agg, n);
        k_edge<<<(E * 32 + 255) / 256, 256, 0, stream>>>(
            h, srcIdx, dstIdx, albf, arbf, dinv, agg, E);
        k_cvt_bf16<<<(nchunks + 255) / 256, 256, 0, stream>>>(agg, aggbf, nchunks);
        dim3 grid(D / BN, (n + BM - 1) / BM);
        k_gemm<<<grid, 256, 0, stream>>>(aggbf, wt, bs + (size_t)l * D, (float*)d_out, n);
        h = (const float*)d_out;   // ping-pong: d_out is next layer's input
    }
}
